// Memory_3693671874651
// MI455X (gfx1250) — matrix operand order repaired and checked by reading, not compiled, not run on hardware
//
#include <hip/hip_runtime.h>
#include <hip/hip_bf16.h>
#include <cstdint>

// ---------------------------------------------------------------------------
// Shapes
// ---------------------------------------------------------------------------
constexpr int Bn = 16, Cc = 512, Hh = 64, Ww = 64;
constexpr int Nn = Bn * Hh * Ww;   // 65536 pixels
constexpr int Mm = 512;            // memory slots
constexpr int CHUNKS = 256;        // column-reduction chunks (Nn/CHUNKS = 256 rows)

typedef __attribute__((ext_vector_type(16))) _Float16 v16h;
typedef __attribute__((ext_vector_type(8)))  _Float16 v8h;
typedef __attribute__((ext_vector_type(8)))  float    v8f;
typedef __attribute__((ext_vector_type(4)))  unsigned int u32x4;
typedef __attribute__((ext_vector_type(8)))  int i32x8;
typedef __attribute__((ext_vector_type(4)))  int i32x4;

// ---------------------------------------------------------------------------
// d_out layout (flat, return order)
// ---------------------------------------------------------------------------
constexpr size_t OUT_UQ   = 0;                                      // (16,1024,64,64)
constexpr size_t OUT_UM   = OUT_UQ + (size_t)Bn * 2 * Cc * Hh * Ww; // 67,108,864
constexpr size_t OUT_SQ   = OUT_UM + (size_t)Mm * Cc;
constexpr size_t OUT_SM   = OUT_SQ + (size_t)Nn * Mm;
constexpr size_t OUT_SEP  = OUT_SM + (size_t)Nn * Mm;
constexpr size_t OUT_COMP = OUT_SEP + 1;

// ---------------------------------------------------------------------------
// workspace layout (byte offsets). Requires ws_size >= ~273 MB.
// ---------------------------------------------------------------------------
constexpr size_t WS_QR     = 0;                                   // N*C f32
constexpr size_t WS_QH     = WS_QR     + (size_t)Nn * Cc * 4;     // N*C f16
constexpr size_t WS_SMH    = WS_QH     + (size_t)Nn * Cc * 2;     // N*C f16
constexpr size_t WS_KH     = WS_SMH    + (size_t)Nn * Cc * 2;     // keys f16
constexpr size_t WS_KTH    = WS_KH     + (size_t)Mm * Cc * 2;     // keys^T f16
constexpr size_t WS_RNORM  = WS_KTH    + (size_t)Mm * Cc * 2;
constexpr size_t WS_RMAX   = WS_RNORM  + (size_t)Nn * 4;
constexpr size_t WS_RINV   = WS_RMAX   + (size_t)Nn * 4;
constexpr size_t WS_T1     = WS_RINV   + (size_t)Nn * 4;
constexpr size_t WS_T2     = WS_T1     + (size_t)Nn * 4;
constexpr size_t WS_PMAX   = WS_T2     + (size_t)Nn * 4;
constexpr size_t WS_PSUM   = WS_PMAX   + (size_t)CHUNKS * Mm * 4;
constexpr size_t WS_CMAX   = WS_PSUM   + (size_t)CHUNKS * Mm * 4;
constexpr size_t WS_CSUM   = WS_CMAX   + 512 * 4;
constexpr size_t WS_CINV   = WS_CSUM   + 512 * 4;
constexpr size_t WS_QUPD   = WS_CINV   + 512 * 4;                 // 512*512 f32

// ---------------------------------------------------------------------------
__global__ void k_zero(float* __restrict__ qupd, float* __restrict__ osep,
                       float* __restrict__ ocomp) {
  int i = blockIdx.x * 256 + threadIdx.x;
  if (i < Mm * Cc) qupd[i] = 0.f;
  if (i == 0) { *osep = 0.f; *ocomp = 0.f; }
}

// per-pixel 1/||q|| (coalesced NCHW reads: lanes vary along W)
__global__ void k_rnorm(const float* __restrict__ q, float* __restrict__ rn) {
  int p = blockIdx.x * 256 + threadIdx.x;
  int b = p >> 12, hw = p & 4095;
  const float* base = q + (size_t)b * Cc * 4096 + hw;
  float ss = 0.f;
  for (int c = 0; c < Cc; ++c) { float x = base[(size_t)c * 4096]; ss += x * x; }
  rn[p] = 1.f / fmaxf(sqrtf(ss), 1e-12f);
}

// NCHW -> (N,C) transpose via LDS tiles; emits qr f32, q f16, out first half
__global__ void k_make_q(const float* __restrict__ q, const float* __restrict__ rn,
                         float* __restrict__ qr, _Float16* __restrict__ qh,
                         float* __restrict__ uq) {
  __shared__ float tile[64][65];
  int p0 = blockIdx.x * 64;
  int b = p0 >> 12, hw0 = p0 & 4095;
  int tid = threadIdx.x;
  for (int cc = 0; cc < Cc; cc += 64) {
    for (int i = tid; i < 64 * 64; i += 256) {
      int c_l = i >> 6, p_l = i & 63;
      float x = q[(size_t)(b * Cc + cc + c_l) * 4096 + hw0 + p_l];
      float v = x * rn[p0 + p_l];
      uq[(size_t)b * (2 * Cc * 4096) + (size_t)(cc + c_l) * 4096 + hw0 + p_l] = v;
      tile[c_l][p_l] = v;
    }
    __syncthreads();
    for (int i = tid; i < 64 * 64; i += 256) {
      int p_l = i >> 6, c_l = i & 63;
      float v = tile[c_l][p_l];
      size_t e = (size_t)(p0 + p_l) * Cc + cc + c_l;
      qr[e] = v;
      qh[e] = (_Float16)v;
    }
    __syncthreads();
  }
}

__global__ void k_keys_cvt(const float* __restrict__ keys,
                           _Float16* __restrict__ kh, _Float16* __restrict__ kth) {
  int e = blockIdx.x * 256 + threadIdx.x;
  int r = e >> 9, c = e & 511;
  float v = keys[e];
  kh[e] = (_Float16)v;
  kth[(size_t)c * 512 + r] = (_Float16)v;
}

// ---------------------------------------------------------------------------
// GEMM: block = 4 waves; block computes 64 rows x 64 cols of D = A(Nx512)B(512x512).
// B panel (64 rows of Brow = 64 cols of B, 64KB f16) is DMA'd to LDS once per
// block via the Tensor Data Mover, then read with ds_load_b128. Each wave holds
// 4 f32 accumulators (16x64) and reuses its A registers across 4 WMMAs.
// Brow[n*512+k] = B[k][n] so each lane's 16 K-values are contiguous.
// mode 0: store row-major (score). mode 1: channel-major into updated_query.
// ---------------------------------------------------------------------------
__global__ __launch_bounds__(128)
void k_gemm(const _Float16* __restrict__ A, const _Float16* __restrict__ Brow,
            float* __restrict__ D0, float* __restrict__ Dq, int mode) {
  __shared__ _Float16 Bs[64 * 512];          // 64 KB B panel, reused as staging
  const int tid  = threadIdx.x;
  const int wv   = tid >> 5;
  const int lane = tid & 31;
  const int l15  = lane & 15;
  const int hi   = lane >> 4;                // 0 | 1
  const int tn   = blockIdx.x * 64;
  const int tm   = (blockIdx.y * 4 + wv) * 16;

#if __has_builtin(__builtin_amdgcn_tensor_load_to_lds)
  if (tid < 32) {                            // wave 0 drives the TDM
    uint64_t ga = (uint64_t)(uintptr_t)(Brow + (size_t)tn * 512);
    uint32_t la = (uint32_t)(uintptr_t)(&Bs[0]);
    u32x4 g0;
    g0[0] = 1u;                                            // count=1 valid D#
    g0[1] = la;                                            // lds_addr
    g0[2] = (uint32_t)ga;                                  // global_addr[31:0]
    g0[3] = (uint32_t)((ga >> 32) & 0x01FFFFFFu) | (2u << 30); // addr[56:32]|type=2
    i32x8 g1;
    g1[0] = 1 << 16;               // workgroup_mask=0, data_size=1 (2 bytes)
    g1[1] = (int)(512u << 16);     // tensor_dim0 = 512 (elements per row)
    g1[2] = (int)(64u << 16);      // tensor_dim1 = 64 rows
    g1[3] = (int)(512u << 16);     // tile_dim0 = 512
    g1[4] = 64;                    // tile_dim1 = 64, tile_dim2 = 0
    g1[5] = 512;                   // tensor_dim0_stride = 512
    g1[6] = 0; g1[7] = 0;
    i32x4 z4 = {0, 0, 0, 0};
#if __clang_major__ >= 23
    i32x8 z8 = {0, 0, 0, 0, 0, 0, 0, 0};
    __builtin_amdgcn_tensor_load_to_lds(g0, g1, z4, z4, z8, 0);
#else
    __builtin_amdgcn_tensor_load_to_lds(g0, g1, z4, z4, 0);
#endif
    __builtin_amdgcn_s_wait_tensorcnt(0);
  }
#else
  for (int i = tid; i < 4096; i += 128)      // 64*512 f16 = 4096 x 16B
    ((ulonglong2*)Bs)[i] = ((const ulonglong2*)(Brow + (size_t)tn * 512))[i];
#endif
  __syncthreads();

  const _Float16* ap  = A + (size_t)(tm + l15) * 512 + hi * 16; // A 16x32 layout, k = 16h + s
  const _Float16* bs0 = Bs + (size_t)l15 * 512 + hi * 16;       // B 32x16 layout
  v8f acc[4] = {{}, {}, {}, {}};
  for (int kk = 0; kk < 512; kk += 32) {
    v8h alo = *(const v8h*)(ap + kk);
    v8h ahi = *(const v8h*)(ap + kk + 8);
    v16h a;
#pragma unroll
    for (int i = 0; i < 8; ++i) { a[i] = alo[i]; a[i + 8] = ahi[i]; }
#pragma unroll
    for (int t = 0; t < 4; ++t) {
      v16h b = *(const v16h*)(bs0 + (size_t)t * 16 * 512 + kk);
      acc[t] = __builtin_amdgcn_wmma_f32_16x16x32_f16(false, a, false, b,
                                                      (short)0, acc[t], false, false);
    }
  }

  if (mode == 0) {
    float* d = D0 + (size_t)(tm + hi * 8) * Mm + tn + l15;
#pragma unroll
    for (int t = 0; t < 4; ++t)
#pragma unroll
      for (int v = 0; v < 8; ++v) d[(size_t)v * Mm + t * 16] = acc[t][v];
  } else {
    __syncthreads();                         // done reading Bs; reuse as staging
    float (*stg)[16][65] = (float (*)[16][65])Bs;  // 4 x 16 x 65 f32 = 16.6 KB
#pragma unroll
    for (int t = 0; t < 4; ++t)
#pragma unroll
      for (int v = 0; v < 8; ++v) stg[wv][hi * 8 + v][t * 16 + l15] = acc[t][v];
    __syncthreads();
    int p = tm + l15;
    int b_ = p >> 12, hw = p & 4095;
    float* d = Dq + (size_t)b_ * (2 * Cc * 4096) + (size_t)(Cc + tn) * 4096 + hw;
#pragma unroll
    for (int t = 0; t < 4; ++t)
#pragma unroll
      for (int j = 0; j < 8; ++j)
        d[(size_t)(t * 16 + hi * 8 + j) * 4096] = stg[wv][l15][t * 16 + hi * 8 + j];
  }
}

// per-row (wave per row): max, 1/sum(exp), top-2 indices
__global__ void k_rowstats(const float* __restrict__ score,
                           float* __restrict__ rmax, float* __restrict__ rinv,
                           int* __restrict__ t1, int* __restrict__ t2) {
  int lane = threadIdx.x & 31;
  int row  = blockIdx.x * 8 + (threadIdx.x >> 5);
  const float* s = score + (size_t)row * Mm;
  float v[16];
#pragma unroll
  for (int j = 0; j < 16; ++j) v[j] = s[lane + 32 * j];
  float m1 = -3.4e38f, m2 = -3.4e38f; int i1 = 0, i2 = 0;
#pragma unroll
  for (int j = 0; j < 16; ++j) {
    float x = v[j]; int idx = lane + 32 * j;
    if (x > m1) { m2 = m1; i2 = i1; m1 = x; i1 = idx; }
    else if (x > m2) { m2 = x; i2 = idx; }
  }
#pragma unroll
  for (int off = 16; off > 0; off >>= 1) {
    float o1 = __shfl_xor(m1, off, 32); int oi1 = __shfl_xor(i1, off, 32);
    float o2 = __shfl_xor(m2, off, 32); int oi2 = __shfl_xor(i2, off, 32);
    if (o1 > m1) {
      if (m1 > o2) { m2 = m1; i2 = i1; } else { m2 = o2; i2 = oi2; }
      m1 = o1; i1 = oi1;
    } else if (o1 > m2) { m2 = o1; i2 = oi1; }
  }
  float se = 0.f;
#pragma unroll
  for (int j = 0; j < 16; ++j) se += __expf(v[j] - m1);
#pragma unroll
  for (int off = 16; off > 0; off >>= 1) se += __shfl_xor(se, off, 32);
  if (lane == 0) { rmax[row] = m1; rinv[row] = 1.f / se; t1[row] = i1; t2[row] = i2; }
}

__global__ void k_colmax_part(const float* __restrict__ score, float* __restrict__ pmax) {
  int ch = blockIdx.x, t = threadIdx.x;
  const float* s = score + (size_t)ch * 256 * Mm;
  float m0 = -3.4e38f, m1 = -3.4e38f;
  for (int r = 0; r < 256; ++r) {
    m0 = fmaxf(m0, s[(size_t)r * Mm + t]);
    m1 = fmaxf(m1, s[(size_t)r * Mm + t + 256]);
  }
  pmax[(size_t)ch * Mm + t] = m0;
  pmax[(size_t)ch * Mm + t + 256] = m1;
}

__global__ void k_colmax_fin(const float* __restrict__ pmax, float* __restrict__ cmax) {
  int c = blockIdx.x * 256 + threadIdx.x;
  float m = -3.4e38f;
  for (int ch = 0; ch < CHUNKS; ++ch) m = fmaxf(m, pmax[(size_t)ch * Mm + c]);
  cmax[c] = m;
}

__global__ void k_colsum_part(const float* __restrict__ score,
                              const float* __restrict__ cmax, float* __restrict__ psum) {
  int ch = blockIdx.x, t = threadIdx.x;
  const float* s = score + (size_t)ch * 256 * Mm;
  float c0 = cmax[t], c1 = cmax[t + 256];
  float s0 = 0.f, s1 = 0.f;
  for (int r = 0; r < 256; ++r) {
    s0 += __expf(s[(size_t)r * Mm + t] - c0);
    s1 += __expf(s[(size_t)r * Mm + t + 256] - c1);
  }
  psum[(size_t)ch * Mm + t] = s0;
  psum[(size_t)ch * Mm + t + 256] = s1;
}

__global__ void k_colsum_fin(const float* __restrict__ psum,
                             float* __restrict__ csum, float* __restrict__ cinv) {
  int c = blockIdx.x * 256 + threadIdx.x;
  float s = 0.f;
  for (int ch = 0; ch < CHUNKS; ++ch) s += psum[(size_t)ch * Mm + c];
  csum[c] = s; cinv[c] = 1.f / s;
}

// score (in s_query slot) -> s_query in place; s_memory (+f16 copy) out
__global__ void k_softmax(float* __restrict__ score, float* __restrict__ smem,
                          _Float16* __restrict__ smh,
                          const float* __restrict__ rmax, const float* __restrict__ rinv,
                          const float* __restrict__ cmax, const float* __restrict__ cinv) {
  size_t e = (size_t)blockIdx.x * 256 + threadIdx.x;
  int p = (int)(e >> 9), j = (int)(e & 511);
  float s = score[e];
  float sm = __expf(s - rmax[p]) * rinv[p];
  smem[e] = sm;
  smh[e] = (_Float16)sm;
  score[e] = __expf(s - cmax[j]) * cinv[j];
}

// losses + weighted segment-sum. w = s_query[p,gi]/colmax(s_query)[gi],
// and colmax(s_query)[j] == 1/colsum[j]  =>  w = s_query[p,gi]*colsum[gi]
__global__ void k_loss(const float* __restrict__ qr, const float* __restrict__ keys,
                       const int* __restrict__ t1, const int* __restrict__ t2,
                       const float* __restrict__ squery, const float* __restrict__ csum,
                       float* __restrict__ qupd, float* __restrict__ osep,
                       float* __restrict__ ocomp) {
  int lane = threadIdx.x & 31;
  int p = blockIdx.x * 8 + (threadIdx.x >> 5);
  int gi = t1[p], ni = t2[p];
  const float* q  = qr + (size_t)p * Cc;
  const float* kp = keys + (size_t)gi * Cc;
  const float* kn = keys + (size_t)ni * Cc;
  float w = squery[(size_t)p * Mm + gi] * csum[gi];
  float sc = 0.f, dp = 0.f, dn = 0.f;
#pragma unroll
  for (int j = 0; j < 16; ++j) {
    int c = lane + 32 * j;
    float qv = q[c];
    float a = qv - kp[c], bb = qv - kn[c];
    sc += a * a;
    float ae = a + 1e-6f, be = bb + 1e-6f;
    dp += ae * ae; dn += be * be;
    atomicAdd(&qupd[(size_t)gi * Cc + c], w * qv);
  }
#pragma unroll
  for (int off = 16; off > 0; off >>= 1) {
    sc += __shfl_xor(sc, off, 32);
    dp += __shfl_xor(dp, off, 32);
    dn += __shfl_xor(dn, off, 32);
  }
  if (lane == 0) {
    atomicAdd(ocomp, sc * (1.f / ((float)Nn * (float)Cc)));
    float sep = sqrtf(dp) - sqrtf(dn) + 1.0f;
    atomicAdd(osep, fmaxf(sep, 0.f) * (1.f / (float)Nn));
  }
}

// updated_memory = l2norm(query_update + keys) per row (wave per row)
__global__ void k_umem(const float* __restrict__ qupd, const float* __restrict__ keys,
                       float* __restrict__ um) {
  int lane = threadIdx.x & 31;
  int m = blockIdx.x * 8 + (threadIdx.x >> 5);
  float v[16]; float ss = 0.f;
#pragma unroll
  for (int j = 0; j < 16; ++j) {
    int c = lane + 32 * j;
    float x = qupd[(size_t)m * Cc + c] + keys[(size_t)m * Cc + c];
    v[j] = x; ss += x * x;
  }
#pragma unroll
  for (int off = 16; off > 0; off >>= 1) ss += __shfl_xor(ss, off, 32);
  float rn = 1.f / fmaxf(sqrtf(ss), 1e-12f);
#pragma unroll
  for (int j = 0; j < 16; ++j) um[(size_t)m * Cc + lane + 32 * j] = v[j] * rn;
}

// ---------------------------------------------------------------------------
extern "C" void kernel_launch(void* const* d_in, const int* in_sizes, int n_in,
                              void* d_out, int out_size, void* d_ws, size_t ws_size,
                              hipStream_t stream) {
  const float* query = (const float*)d_in[0];
  const float* keys  = (const float*)d_in[1];
  float* out = (float*)d_out;
  char* ws = (char*)d_ws;

  float*    qr    = (float*)   (ws + WS_QR);
  _Float16* qh    = (_Float16*)(ws + WS_QH);
  _Float16* smh   = (_Float16*)(ws + WS_SMH);
  _Float16* kh    = (_Float16*)(ws + WS_KH);
  _Float16* kth   = (_Float16*)(ws + WS_KTH);
  float*    rnorm = (float*)(ws + WS_RNORM);
  float*    rmax  = (float*)(ws + WS_RMAX);
  float*    rinv  = (float*)(ws + WS_RINV);
  int*      t1    = (int*)  (ws + WS_T1);
  int*      t2    = (int*)  (ws + WS_T2);
  float*    pmax  = (float*)(ws + WS_PMAX);
  float*    psum  = (float*)(ws + WS_PSUM);
  float*    cmax  = (float*)(ws + WS_CMAX);
  float*    csum  = (float*)(ws + WS_CSUM);
  float*    cinv  = (float*)(ws + WS_CINV);
  float*    qupd  = (float*)(ws + WS_QUPD);

  float* uq    = out + OUT_UQ;
  float* um    = out + OUT_UM;
  float* score = out + OUT_SQ;   // raw scores staged here, overwritten with s_query
  float* smem  = out + OUT_SM;
  float* osep  = out + OUT_SEP;
  float* ocomp = out + OUT_COMP;

  k_zero<<<(Mm * Cc + 255) / 256, 256, 0, stream>>>(qupd, osep, ocomp);
  k_rnorm<<<Nn / 256, 256, 0, stream>>>(query, rnorm);
  k_make_q<<<Nn / 64, 256, 0, stream>>>(query, rnorm, qr, qh, uq);
  k_keys_cvt<<<(Mm * Cc) / 256, 256, 0, stream>>>(keys, kh, kth);

  // score = qr @ keys^T  (B[k][n] = keys[n][k] -> contiguous rows = kh)
  k_gemm<<<dim3(Mm / 64, Nn / 64), 128, 0, stream>>>(qh, kh, score, nullptr, 0);

  k_rowstats<<<Nn / 8, 256, 0, stream>>>(score, rmax, rinv, t1, t2);
  k_colmax_part<<<CHUNKS, 256, 0, stream>>>(score, pmax);
  k_colmax_fin<<<Mm / 256, 256, 0, stream>>>(pmax, cmax);
  k_colsum_part<<<CHUNKS, 256, 0, stream>>>(score, cmax, psum);
  k_colsum_fin<<<Mm / 256, 256, 0, stream>>>(psum, csum, cinv);
  k_softmax<<<(size_t)Nn * Mm / 256, 256, 0, stream>>>(score, smem, smh,
                                                       rmax, rinv, cmax, cinv);

  // concat = s_memory @ keys  (B[k][n] = keys[k][n] -> contiguous rows = kth)
  k_gemm<<<dim3(Mm / 64, Nn / 64), 128, 0, stream>>>(smh, kth, nullptr, uq, 1);

  k_loss<<<Nn / 8, 256, 0, stream>>>(qr, keys, t1, t2, score, csum,
                                     qupd, osep, ocomp);
  k_umem<<<Mm / 8, 256, 0, stream>>>(qupd, keys, um);
}
